// MultiHeadAttention_6682969113138
// MI455X (gfx1250) — matrix operand order repaired and checked by reading, not compiled, not run on hardware
//
#include <hip/hip_runtime.h>
#include <hip/hip_bf16.h>

// MHA forward: B=2, S=2048, E=1024, H=16, D=64. All GEMMs via v_wmma_f32_16x16x32_bf16.
// GEMM tiles are filled with GLOBAL_LOAD_ASYNC_TO_LDS_B128 (ASYNCcnt) + double buffering.

#define Bc 2
#define Sc 2048
#define Ec 1024
#define Hc 16
#define Dc 64
#define BSc (Bc * Sc)   // 4096 rows

typedef __attribute__((ext_vector_type(16))) __bf16 bf16x16;
typedef __attribute__((ext_vector_type(8)))  float  f32x8;

union Frag {
    uint4        q[2];
    unsigned int u[8];
    bf16x16      v;
};

__device__ __forceinline__ unsigned short f32_to_bf16(float f) {
    unsigned int u = __float_as_uint(f);
    unsigned int r = u + 0x7FFFu + ((u >> 16) & 1u);  // round-to-nearest-even
    return (unsigned short)(r >> 16);
}

__device__ __forceinline__ f32x8 zero8() {
    f32x8 v;
#pragma unroll
    for (int i = 0; i < 8; ++i) v[i] = 0.0f;
    return v;
}

__device__ __forceinline__ f32x8 wmma_bf16(const Frag& a, const Frag& b, f32x8 c) {
    // (neg_a, A, neg_b, B, c_mod, C, reuse_a, reuse_b)
    return __builtin_amdgcn_wmma_f32_16x16x32_bf16(false, a.v, false, b.v,
                                                   (short)0, c, false, false);
}

// Async DMA: 16 bytes per lane, global -> LDS, tracked by ASYNCcnt (no VGPR staging).
__device__ __forceinline__ void async_load_b128(unsigned lds_off, const void* gaddr) {
    asm volatile("global_load_async_to_lds_b128 %0, %1, off"
                 :: "v"(lds_off), "v"((unsigned long long)(size_t)gaddr)
                 : "memory");
}
__device__ __forceinline__ void wait_async0() {
    asm volatile("s_wait_asynccnt 0" ::: "memory");
}

// ---------------------------------------------------------------- pack kernels

__global__ void pack_bf16_kernel(const float* __restrict__ in,
                                 unsigned short* __restrict__ out, int n) {
    int i = blockIdx.x * blockDim.x + threadIdx.x;
    if (i < n) out[i] = f32_to_bf16(in[i]);
}

// out[n*E + k] = bf16(in[k*E + n]); 32x32 LDS-tiled, coalesced both sides.
__global__ __launch_bounds__(256) void transpose_bf16_kernel(
    const float* __restrict__ in, unsigned short* __restrict__ out) {
    __shared__ float T[32][33];   // pad to avoid bank conflicts
    const int tid = threadIdx.x;
    const int k0 = blockIdx.x * 32;   // source row block (k)
    const int n0 = blockIdx.y * 32;   // source col block (n)
#pragma unroll
    for (int i = 0; i < 4; ++i) {
        int c2 = tid + i * 256;       // 0..1023
        int r = c2 >> 5, c = c2 & 31;
        T[r][c] = in[(size_t)(k0 + r) * Ec + n0 + c];
    }
    __syncthreads();
#pragma unroll
    for (int i = 0; i < 4; ++i) {
        int c2 = tid + i * 256;
        int r = c2 >> 5, c = c2 & 31;   // output row n0+r, col k0+c
        out[(size_t)(n0 + r) * Ec + k0 + c] = f32_to_bf16(T[c][r]);
    }
}

// ---------------------------------------------------------------- GEMM (bf16 -> fp32 acc)
// C[M=4096, N=1024] = A[M,E] * B[E,N] + bias, B given pre-transposed as Bt[N,E].
// MODE 0: write bf16 head-split [B,H,S,D]   (Q and K)
// MODE 2: write bf16 transposed [B,H,D,S]   (V^T)
// MODE 3: write fp32 row-major [M,N]        (final output)
template <int MODE>
__global__ __launch_bounds__(256) void gemm_bf16_kernel(
    const unsigned short* __restrict__ A,
    const unsigned short* __restrict__ Bt,
    const float* __restrict__ bias,
    void* __restrict__ out) {
    __shared__ __align__(16) unsigned short Xs[2][128 * 32];  // double-buffered
    __shared__ __align__(16) unsigned short Ws[2][128 * 32];

    const int tid  = threadIdx.x;
    const int lane = tid & 31;
    const int w    = tid >> 5;       // 8 waves
    const int half = lane >> 4;
    const int l16  = lane & 15;
    const int bm   = blockIdx.y;     // 32 row blocks of 128
    const int bn   = blockIdx.x;     // 8  col blocks of 128
    const int wm   = w >> 1;         // 0..3 : 32-row strip
    const int wn   = w & 1;          // 0..1 : 64-col strip

    // this thread's two 16B chunks per tile
    const int c0 = tid, c1 = tid + 256;
    const int r0 = c0 >> 2, kc0 = (c0 & 3) << 3;
    const int r1 = c1 >> 2, kc1 = (c1 & 3) << 3;
    const unsigned short* Ar0 = &A[(size_t)(bm * 128 + r0) * Ec + kc0];
    const unsigned short* Ar1 = &A[(size_t)(bm * 128 + r1) * Ec + kc1];
    const unsigned short* Br0 = &Bt[(size_t)(bn * 128 + r0) * Ec + kc0];
    const unsigned short* Br1 = &Bt[(size_t)(bn * 128 + r1) * Ec + kc1];

    f32x8 acc[2][4];
#pragma unroll
    for (int rt = 0; rt < 2; ++rt)
#pragma unroll
        for (int ct = 0; ct < 4; ++ct) acc[rt][ct] = zero8();

    const int T = Ec / 32;  // 32 K-tiles

    // prologue: async-fill tile 0 into buffer 0
    async_load_b128((unsigned)(size_t)&Xs[0][r0 * 32 + kc0], Ar0);
    async_load_b128((unsigned)(size_t)&Xs[0][r1 * 32 + kc1], Ar1);
    async_load_b128((unsigned)(size_t)&Ws[0][r0 * 32 + kc0], Br0);
    async_load_b128((unsigned)(size_t)&Ws[0][r1 * 32 + kc1], Br1);

    for (int t = 0; t < T; ++t) {
        wait_async0();       // my fills for the current buffer are done
        __syncthreads();     // everyone's fills visible; prior reads of next buffer done

        const int cur = t & 1;
        if (t + 1 < T) {     // async-fill next tile into the other buffer
            const int nb = (t + 1) & 1;
            const int ko = (t + 1) * 32;
            async_load_b128((unsigned)(size_t)&Xs[nb][r0 * 32 + kc0], Ar0 + ko);
            async_load_b128((unsigned)(size_t)&Xs[nb][r1 * 32 + kc1], Ar1 + ko);
            async_load_b128((unsigned)(size_t)&Ws[nb][r0 * 32 + kc0], Br0 + ko);
            async_load_b128((unsigned)(size_t)&Ws[nb][r1 * 32 + kc1], Br1 + ko);
        }

        Frag af[2], bf[4];
        const int kb  = half * 8;    // A-fragment K sub-base
        const int kbb = half * 8;    // B-fragment K sub-base
#pragma unroll
        for (int rt = 0; rt < 2; ++rt) {
            int r = wm * 32 + rt * 16 + l16;
            af[rt].q[0] = *(const uint4*)&Xs[cur][r * 32 + kb];
            af[rt].q[1] = *(const uint4*)&Xs[cur][r * 32 + 16 + kb];
        }
#pragma unroll
        for (int ct = 0; ct < 4; ++ct) {
            int n = wn * 64 + ct * 16 + l16;
            bf[ct].q[0] = *(const uint4*)&Ws[cur][n * 32 + kbb];
            bf[ct].q[1] = *(const uint4*)&Ws[cur][n * 32 + kbb + 16];
        }
#pragma unroll
        for (int rt = 0; rt < 2; ++rt)
#pragma unroll
            for (int ct = 0; ct < 4; ++ct)
                acc[rt][ct] = wmma_bf16(af[rt], bf[ct], acc[rt][ct]);
    }

    // epilogue
#pragma unroll
    for (int rt = 0; rt < 2; ++rt)
#pragma unroll
        for (int ct = 0; ct < 4; ++ct)
#pragma unroll
            for (int j = 0; j < 8; ++j) {
                int gr = bm * 128 + wm * 32 + rt * 16 + j + 8 * half;
                int gc = bn * 128 + wn * 64 + ct * 16 + l16;
                float v = acc[rt][ct][j] + bias[gc];
                if (MODE == 3) {
                    ((float*)out)[(size_t)gr * Ec + gc] = v;
                } else {
                    int b = gr / Sc, s = gr % Sc;
                    int h = gc / Dc, d = gc % Dc;
                    unsigned short bv = f32_to_bf16(v);
                    if (MODE == 2)
                        ((unsigned short*)out)[(((size_t)(b * Hc + h)) * Dc + d) * Sc + s] = bv;
                    else
                        ((unsigned short*)out)[(((size_t)(b * Hc + h)) * Sc + s) * Dc + d] = bv;
                }
            }
}

// ---------------------------------------------------------------- flash attention
// Q,K: bf16 [B,H,S,D]; Vt: bf16 [B,H,D,S]; out Xattn: bf16 [B,S,E] (heads merged).
// One wave owns 16 query rows; causal loop length varies per wave (no block sync).
__global__ __launch_bounds__(256) void attn_kernel(
    const unsigned short* __restrict__ Q,
    const unsigned short* __restrict__ K,
    const unsigned short* __restrict__ Vt,
    unsigned short* __restrict__ Xattn) {
    __shared__ __align__(16) unsigned short Pl[8 * 16 * 32];  // per-wave 16x32 P patch

    const int tid  = threadIdx.x;
    const int lane = tid & 31;
    const int w    = tid >> 5;
    const int half = lane >> 4;
    const int l16  = lane & 15;
    const int bh   = blockIdx.y;                 // B*H = 32
    const int q0   = blockIdx.x * 128 + w * 16;  // wave's query tile base

    const unsigned short* Qh = Q  + (size_t)bh * Sc * Dc;
    const unsigned short* Kh = K  + (size_t)bh * Sc * Dc;
    const unsigned short* Vh = Vt + (size_t)bh * Dc * Sc;
    unsigned short* Pw = &Pl[w * (16 * 32)];

    // Q A-fragments (d 0..31 and 32..63), loaded once
    Frag qf[2];
    {
        const unsigned short* qr = Qh + (size_t)(q0 + l16) * Dc;
        int kb = half * 8;
        qf[0].q[0] = *(const uint4*)(qr + kb);
        qf[0].q[1] = *(const uint4*)(qr + 16 + kb);
        qf[1].q[0] = *(const uint4*)(qr + 32 + kb);
        qf[1].q[1] = *(const uint4*)(qr + 48 + kb);
    }

    float m[8], l[8];
    f32x8 o[4];
#pragma unroll
    for (int j = 0; j < 8; ++j) { m[j] = -3.0e38f; l[j] = 0.0f; }
#pragma unroll
    for (int dt = 0; dt < 4; ++dt) o[dt] = zero8();

    const int nk32 = (q0 + 16 + 31) >> 5;  // causal: keys 0 .. q0+15
    for (int t32 = 0; t32 < nk32; ++t32) {
        const int kb32 = t32 * 32;
        f32x8 s[2];

        if (t32 + 1 < nk32) {  // speculative L2 prefetch of next key block
            __builtin_prefetch(Kh + (size_t)(kb32 + 32 + l16) * Dc, 0, 1);
            __builtin_prefetch(Vh + (size_t)l16 * Sc + kb32 + 32, 0, 1);
        }

        // ---- load ALL K fragments first (one load clause, one wait, 4 wmma)
        Frag kf[2][2];
        {
            const int kbb = half * 8;
#pragma unroll
            for (int th = 0; th < 2; ++th) {
                const unsigned short* kr = Kh + (size_t)(kb32 + th * 16 + l16) * Dc;
                kf[th][0].q[0] = *(const uint4*)(kr + kbb);
                kf[th][0].q[1] = *(const uint4*)(kr + kbb + 16);
                kf[th][1].q[0] = *(const uint4*)(kr + 32 + kbb);
                kf[th][1].q[1] = *(const uint4*)(kr + 32 + kbb + 16);
            }
        }
#pragma unroll
        for (int th = 0; th < 2; ++th) {
            f32x8 sc_ = zero8();
            sc_ = wmma_bf16(qf[0], kf[th][0], sc_);
            sc_ = wmma_bf16(qf[1], kf[th][1], sc_);
            s[th] = sc_;
        }

        // ---- issue V fragment loads NOW; their latency hides under softmax + LDS wait
        Frag vf[4];
#pragma unroll
        for (int dt = 0; dt < 4; ++dt) {
            const unsigned short* vr =
                Vh + (size_t)(dt * 16 + l16) * Sc + kb32 + half * 8;
            vf[dt].q[0] = *(const uint4*)vr;
            vf[dt].q[1] = *(const uint4*)(vr + 16);
        }

        // ---- scale 1/sqrt(D) + causal mask (exact -10000 like reference)
#pragma unroll
        for (int th = 0; th < 2; ++th)
#pragma unroll
            for (int j = 0; j < 8; ++j) {
                int kg = kb32 + th * 16 + l16;
                int rg = q0 + j + 8 * half;
                s[th][j] = (kg <= rg) ? s[th][j] * 0.125f : -10000.0f;
            }

        // ---- online softmax (row stats live across a 16-lane half)
        float nm[8], rescale[8];
#pragma unroll
        for (int j = 0; j < 8; ++j) {
            float mx = fmaxf(s[0][j], s[1][j]);
#pragma unroll
            for (int off = 8; off; off >>= 1) mx = fmaxf(mx, __shfl_xor(mx, off, 16));
            nm[j] = fmaxf(m[j], mx);
        }
#pragma unroll
        for (int j = 0; j < 8; ++j) {
            float p0 = __expf(s[0][j] - nm[j]);
            float p1 = __expf(s[1][j] - nm[j]);
            s[0][j] = p0; s[1][j] = p1;
            float rs = p0 + p1;
#pragma unroll
            for (int off = 8; off; off >>= 1) rs += __shfl_xor(rs, off, 16);
            rescale[j] = __expf(m[j] - nm[j]);
            l[j] = l[j] * rescale[j] + rs;
            m[j] = nm[j];
        }
#pragma unroll
        for (int dt = 0; dt < 4; ++dt)
#pragma unroll
            for (int j = 0; j < 8; ++j) o[dt][j] *= rescale[j];

        // ---- C-layout P -> LDS -> A-layout fragment
#pragma unroll
        for (int j = 0; j < 8; ++j) {
            int r = j + 8 * half;
            Pw[r * 32 + l16]      = f32_to_bf16(s[0][j]);
            Pw[r * 32 + 16 + l16] = f32_to_bf16(s[1][j]);
        }
        __asm__ volatile("s_wait_dscnt 0" ::: "memory");
        Frag pf;
        {
            int kb = half * 8;
            pf.q[0] = *(const uint4*)&Pw[l16 * 32 + kb];
            pf.q[1] = *(const uint4*)&Pw[l16 * 32 + 16 + kb];
        }
        __asm__ volatile("" ::: "memory");

        // ---- O += P @ V  (4 d-tiles, contraction over 32 keys)
#pragma unroll
        for (int dt = 0; dt < 4; ++dt) o[dt] = wmma_bf16(pf, vf[dt], o[dt]);
    }

    // ---- normalize and write merged-head bf16 activation [B,S,E]
    const int b = bh >> 4;   // /Hc
    const int h = bh & 15;   // %Hc
#pragma unroll
    for (int j = 0; j < 8; ++j) {
        float inv = 1.0f / l[j];
        int srow = q0 + j + 8 * half;
#pragma unroll
        for (int dt = 0; dt < 4; ++dt) {
            int d = dt * 16 + l16;
            Xattn[((size_t)b * Sc + srow) * Ec + h * Dc + d] = f32_to_bf16(o[dt][j] * inv);
        }
    }
}

// ---------------------------------------------------------------- launch

extern "C" void kernel_launch(void* const* d_in, const int* in_sizes, int n_in,
                              void* d_out, int out_size, void* d_ws, size_t ws_size,
                              hipStream_t stream) {
    const float* x  = (const float*)d_in[0];
    const float* Wq = (const float*)d_in[1];
    const float* bq = (const float*)d_in[2];
    const float* Wk = (const float*)d_in[3];
    const float* bk = (const float*)d_in[4];
    const float* Wv = (const float*)d_in[5];
    const float* bv = (const float*)d_in[6];
    const float* Wo = (const float*)d_in[7];
    const float* bo = (const float*)d_in[8];

    char* ws = (char*)d_ws;
    const size_t MB = 1ull << 20;
    unsigned short* Xbf   = (unsigned short*)(ws + 0 * MB);   // [4096,1024] bf16 (8 MB)
    unsigned short* Wqt   = (unsigned short*)(ws + 8 * MB);   // [1024,1024] bf16^T (2 MB)
    unsigned short* Wkt   = (unsigned short*)(ws + 10 * MB);
    unsigned short* Wvt   = (unsigned short*)(ws + 12 * MB);
    unsigned short* Wot   = (unsigned short*)(ws + 14 * MB);
    unsigned short* Qb    = (unsigned short*)(ws + 16 * MB);  // [B,H,S,D] bf16 (8 MB)
    unsigned short* Kb    = (unsigned short*)(ws + 24 * MB);  // [B,H,S,D]
    unsigned short* Vtb   = (unsigned short*)(ws + 32 * MB);  // [B,H,D,S]
    unsigned short* Xattn = (unsigned short*)(ws + 40 * MB);  // [B,S,E] bf16

    // 1) pack activations + transpose-pack weights to bf16
    pack_bf16_kernel<<<(BSc * Ec + 255) / 256, 256, 0, stream>>>(x, Xbf, BSc * Ec);
    dim3 tgrid(Ec / 32, Ec / 32);
    transpose_bf16_kernel<<<tgrid, 256, 0, stream>>>(Wq, Wqt);
    transpose_bf16_kernel<<<tgrid, 256, 0, stream>>>(Wk, Wkt);
    transpose_bf16_kernel<<<tgrid, 256, 0, stream>>>(Wv, Wvt);
    transpose_bf16_kernel<<<tgrid, 256, 0, stream>>>(Wo, Wot);

    // 2) QKV projections (V written transposed)
    dim3 ggrid(Ec / 128, BSc / 128);  // (8, 32)
    gemm_bf16_kernel<0><<<ggrid, 256, 0, stream>>>(Xbf, Wqt, bq, Qb);
    gemm_bf16_kernel<0><<<ggrid, 256, 0, stream>>>(Xbf, Wkt, bk, Kb);
    gemm_bf16_kernel<2><<<ggrid, 256, 0, stream>>>(Xbf, Wvt, bv, Vtb);

    // 3) causal flash attention
    attn_kernel<<<dim3(Sc / 128, Bc * Hc), 256, 0, stream>>>(Qb, Kb, Vtb, Xattn);

    // 4) output projection -> fp32 d_out
    gemm_bf16_kernel<3><<<ggrid, 256, 0, stream>>>(Xattn, Wot, bo, (float*)d_out);
}
